// VariationalAutoEncoder_73753178406915
// MI455X (gfx1250) — compile-verified, matrix operand order checked by reading, not run on hardware
//
#include <hip/hip_runtime.h>
#include <hip/hip_bf16.h>
#include <math.h>

// ---------------- problem constants ----------------
#define N_NODES 100000
#define N_EDGES 1600000
#define NBATCH  512
#define DIN     64
#define HDIM    128
#define LATD    32
#define HDEC    256
#define NMAXV   50
#define NPAIRV  1225          // 50*49/2
#define NOUTV   (2*NPAIRV)    // 2450
#define BN_RS   0.9999950000374997f   // rsqrt(1 + 1e-5)

typedef __attribute__((ext_vector_type(16))) _Float16 v16h;
typedef __attribute__((ext_vector_type(8)))  _Float16 v8h;
typedef __attribute__((ext_vector_type(8)))  float    v8f;

// ---------------------------------------------------------------------------
// Fused GEMM: C[M,N] = epi(A[M,K] @ W[K,N] + bias)
//   epi: optional PReLU(alpha) then optional affine-BN (g*rsqrt(1+eps), b)
// Requirements: M % 16 == 0 (true for all call sites), K compile-time mult of 32.
// Block: 256 threads = 8 wave32, covering a 64-row x 128-col output panel
// (4 M-tiles per wave). W panel is staged in LDS transposed to [col][k] (f16)
// so B fragments are two contiguous ds_load_b128 per K-step; A fragments are
// four unconditional global_load_b128 per K-step. No branches in hot loop.
// ---------------------------------------------------------------------------
template<int K, bool PRELU, bool BN>
__global__ __launch_bounds__(256)
void gemm_wmma(const float* __restrict__ A,
               const float* __restrict__ W,
               const float* __restrict__ bias,
               float* __restrict__ C,
               int M, int N,
               const float* __restrict__ prelu_a,
               const float* __restrict__ bn_g,
               const float* __restrict__ bn_b)
{
    __shared__ _Float16 ldsW[128 * K];       // [col][k], <= 64 KB (K=256)

    const int tilesM  = M >> 4;
    const int colBase = blockIdx.x << 7;     // 128 output columns per block

    // Cooperative W panel load: coalesced f32 reads, f16 convert, transpose.
    // Column index clamped (ragged N=2450); clamped columns are never stored.
    for (int idx = threadIdx.x; idx < 128 * K; idx += 256) {
        const int col = idx & 127;           // consecutive tid -> consecutive col
        const int k   = idx >> 7;
        int cg = colBase + col;
        cg = (cg < N) ? cg : (N - 1);
        ldsW[col * K + k] = (_Float16)W[(size_t)k * N + cg];
    }
    __syncthreads();

    const int waveId = threadIdx.x >> 5;     // 0..7: column tile within panel
    const int lane   = threadIdx.x & 31;
    const int half   = lane >> 4;            // 0: lanes 0-15, 1: lanes 16-31
    const int lr     = lane & 15;
    const int n      = colBase + (waveId << 4) + lr;   // this lane's output col
    const bool nOK   = (n < N);

    // Column-constant epilogue params (hoisted; n fixed across all rows)
    const float cb    = nOK ? bias[n] : 0.0f;
    const float csc   = (BN && nOK) ? bn_g[n] * BN_RS : 1.0f;
    const float csh   = (BN && nOK) ? bn_b[n] : 0.0f;
    const float alpha = PRELU ? *prelu_a : 0.0f;

    // This lane's B column in LDS
    const _Float16* wcol = &ldsW[((waveId << 4) + lr) * K];

    for (int mi = 0; mi < 4; ++mi) {
        const int mT = (blockIdx.y << 2) + mi;
        if (mT >= tilesM) break;
        const int mA = (mT << 4) + lr;       // A row this lane supplies
        const float* arow = A + (size_t)mA * K;
        if (mT + 1 < tilesM)
            __builtin_prefetch(arow + (size_t)16 * K, 0, 0);  // next M-tile

        v8f acc = {};
#pragma unroll
        for (int kk = 0; kk < K; kk += 32) {
            // fragment element j holds k = kk + (j<8 ? j : j+8) + half*8:
            // two contiguous 8-element runs at o and o+16.
            const int o = kk + (half << 3);
            const float4 a0 = *(const float4*)(arow + o);
            const float4 a1 = *(const float4*)(arow + o + 4);
            const float4 a2 = *(const float4*)(arow + o + 16);
            const float4 a3 = *(const float4*)(arow + o + 20);
            const v8h bl = *(const v8h*)(wcol + o);        // ds_load_b128
            const v8h bh = *(const v8h*)(wcol + o + 16);   // ds_load_b128

            v16h a, b;
            a[0]  = (_Float16)a0.x; a[1]  = (_Float16)a0.y;
            a[2]  = (_Float16)a0.z; a[3]  = (_Float16)a0.w;
            a[4]  = (_Float16)a1.x; a[5]  = (_Float16)a1.y;
            a[6]  = (_Float16)a1.z; a[7]  = (_Float16)a1.w;
            a[8]  = (_Float16)a2.x; a[9]  = (_Float16)a2.y;
            a[10] = (_Float16)a2.z; a[11] = (_Float16)a2.w;
            a[12] = (_Float16)a3.x; a[13] = (_Float16)a3.y;
            a[14] = (_Float16)a3.z; a[15] = (_Float16)a3.w;
#pragma unroll
            for (int j = 0; j < 8; ++j) { b[j] = bl[j]; b[j + 8] = bh[j]; }

            acc = __builtin_amdgcn_wmma_f32_16x16x32_f16(false, a, false, b,
                                                         (short)0, acc, false, false);
        }

        // C/D layout: VGPR r -> row r (+8 for hi lanes), column = lane
        if (nOK) {
            const int mBase = (mT << 4) + (half << 3);
#pragma unroll
            for (int r = 0; r < 8; ++r) {
                float v = acc[r] + cb;
                if (PRELU) v = (v > 0.0f) ? v : alpha * v;
                if (BN)    v = v * csc + csh;
                C[(size_t)(mBase + r) * N + n] = v;
            }
        }
    }
}

// ---------------- helpers ----------------
__global__ void k_copy_f4(float4* __restrict__ dst, const float4* __restrict__ src, int n4)
{
    int i = blockIdx.x * blockDim.x + threadIdx.x;
    int stride = gridDim.x * blockDim.x;
    for (; i < n4; i += stride) dst[i] = src[i];
}

__global__ void k_zero(float* __restrict__ p, int n)
{
    int i = blockIdx.x * blockDim.x + threadIdx.x;
    int stride = gridDim.x * blockDim.x;
    for (; i < n; i += stride) p[i] = 0.0f;
}

// z[dst] += h[src] over all edges. One wave per edge; lane handles float4 (H=128 = 32*4).
__global__ void k_edge_agg(const float* __restrict__ h, const int* __restrict__ ei,
                           float* __restrict__ z, int nE)
{
    const int wid  = (blockIdx.x * blockDim.x + threadIdx.x) >> 5;
    const int lane = threadIdx.x & 31;
    if (wid >= nE) return;
    const int src = ei[wid];
    const int dst = ei[nE + wid];
    const float4 v = *(const float4*)(h + (size_t)src * HDIM + lane * 4);
    float* zp = z + (size_t)dst * HDIM + lane * 4;
    atomicAdd(zp + 0, v.x);
    atomicAdd(zp + 1, v.y);
    atomicAdd(zp + 2, v.z);
    atomicAdd(zp + 3, v.w);
}

// g[batch[n]] += h[n]. One wave per node.
__global__ void k_pool(const float* __restrict__ h, const int* __restrict__ batch,
                       float* __restrict__ g, int nNodes)
{
    const int wid  = (blockIdx.x * blockDim.x + threadIdx.x) >> 5;
    const int lane = threadIdx.x & 31;
    if (wid >= nNodes) return;
    const int bi = batch[wid];
    const float4 v = *(const float4*)(h + (size_t)wid * HDIM + lane * 4);
    float* gp = g + (size_t)bi * HDIM + lane * 4;
    atomicAdd(gp + 0, v.x);
    atomicAdd(gp + 1, v.y);
    atomicAdd(gp + 2, v.z);
    atomicAdd(gp + 3, v.w);
}

// x = x * (g[c]*rsqrt(1+eps)) + b[c], rows of width C
__global__ void k_bn_rows(float* __restrict__ x, const float* __restrict__ g,
                          const float* __restrict__ b, int n, int C)
{
    int i = blockIdx.x * blockDim.x + threadIdx.x;
    if (i < n) {
        int c = i % C;
        x[i] = x[i] * (g[c] * BN_RS) + b[c];
    }
}

// x_g = mu + eps * exp(0.5*logvar)
__global__ void k_reparam(const float* __restrict__ mu, const float* __restrict__ lv,
                          const float* __restrict__ eps, float* __restrict__ xg, int n)
{
    int i = blockIdx.x * blockDim.x + threadIdx.x;
    if (i < n) xg[i] = mu[i] + eps[i] * expf(0.5f * lv[i]);
}

// Straight-through Gumbel argmax -> symmetric adjacency (every element written)
__global__ void k_adj(const float* __restrict__ logits, const float* __restrict__ gum,
                      float* __restrict__ adj)
{
    int i = blockIdx.x * blockDim.x + threadIdx.x;
    if (i >= NBATCH * NMAXV * NMAXV) return;
    const int b   = i / (NMAXV * NMAXV);
    const int rem = i % (NMAXV * NMAXV);
    const int r = rem / NMAXV;
    const int c = rem % NMAXV;
    float v = 0.0f;
    if (r != c) {
        const int lo = (r < c) ? r : c;
        const int hi = (r < c) ? c : r;
        const int p  = lo * (NMAXV - 1) - (lo * (lo - 1)) / 2 + (hi - lo - 1);
        const size_t off = ((size_t)b * NPAIRV + p) * 2;
        const float s0 = logits[off + 0] + gum[off + 0];
        const float s1 = logits[off + 1] + gum[off + 1];
        v = (s0 >= s1) ? 1.0f : 0.0f;   // argmax ties -> index 0
    }
    adj[i] = v;
}

// ---------------- host-side launcher ----------------
template<int K, bool PRELU, bool BN>
static inline void launch_gemm(const float* A, const float* W, const float* bias,
                               float* C, int M, int N,
                               const float* alpha, const float* bg, const float* bb,
                               hipStream_t s)
{
    dim3 grid((N + 127) / 128, ((M >> 4) + 3) / 4);
    gemm_wmma<K, PRELU, BN><<<grid, 256, 0, s>>>(A, W, bias, C, M, N, alpha, bg, bb);
}

// Input leaf order = JAX pytree flatten (dict keys sorted, lists in order):
// 0 bn_b, 1 bn_g,
// 2+10l .. conv l: a1,a2,b1,b2,be1,bnb,bng,g1,w1,w2   (l = 0..2)
// 32 dec0_b, 33 dec0_w, 34 dec1_b, 35 dec1_w, 36 dec2_b, 37 dec2_w,
// 38 dec_a, 39 fc0_b, 40 fc0_w, 41 fc_b, 42 fc_w,
// 43 lv_b, 44 lv_w, 45 mu_b, 46 mu_w,
// 47 x, 48 eps, 49 gumbel, 50 edge_index, 51 batch
extern "C" void kernel_launch(void* const* d_in, const int* in_sizes, int n_in,
                              void* d_out, int out_size, void* d_ws, size_t ws_size,
                              hipStream_t stream)
{
    (void)in_sizes; (void)n_in; (void)out_size; (void)ws_size;

    const float* bn_b   = (const float*)d_in[0];
    const float* bn_g   = (const float*)d_in[1];
    const float* dec0_b = (const float*)d_in[32];
    const float* dec0_w = (const float*)d_in[33];
    const float* dec1_b = (const float*)d_in[34];
    const float* dec1_w = (const float*)d_in[35];
    const float* dec2_b = (const float*)d_in[36];
    const float* dec2_w = (const float*)d_in[37];
    const float* dec_a  = (const float*)d_in[38];
    const float* fc0_b  = (const float*)d_in[39];
    const float* fc0_w  = (const float*)d_in[40];
    const float* fc_b   = (const float*)d_in[41];
    const float* fc_w   = (const float*)d_in[42];
    const float* lv_b   = (const float*)d_in[43];
    const float* lv_w   = (const float*)d_in[44];
    const float* mu_b   = (const float*)d_in[45];
    const float* mu_w   = (const float*)d_in[46];
    const float* x      = (const float*)d_in[47];
    const float* eps    = (const float*)d_in[48];
    const float* gumbel = (const float*)d_in[49];
    const int*   edges  = (const int*)d_in[50];
    const int*   batch  = (const int*)d_in[51];

    // workspace carve-up (floats)
    float* ws    = (float*)d_ws;
    float* buf0  = ws;                                   // 12.8M
    float* buf1  = buf0 + (size_t)N_NODES * HDIM;        // 12.8M
    float* gbuf  = buf1 + (size_t)N_NODES * HDIM;        // 512*128
    float* g2    = gbuf + (size_t)NBATCH * HDIM;         // 512*128
    float* dbuf1 = g2   + (size_t)NBATCH * HDIM;         // 512*256
    float* dbuf2 = dbuf1 + (size_t)NBATCH * HDEC;        // 512*256
    float* logit = dbuf2 + (size_t)NBATCH * HDEC;        // 512*2450

    float* out     = (float*)d_out;
    float* out_adj = out;                                // 512*50*50
    float* out_mu  = out + (size_t)NBATCH * NMAXV * NMAXV;
    float* out_lv  = out_mu + (size_t)NBATCH * LATD;
    float* out_xg  = out_lv + (size_t)NBATCH * LATD;

    // 1) h = x @ fc0_w + fc0_b   (100000 x 64 x 128)
    launch_gemm<DIN, false, false>(x, fc0_w, fc0_b, buf0, N_NODES, HDIM,
                                   nullptr, nullptr, nullptr, stream);

    // 2) three GraphConv layers, ping-pong between buf0/buf1
    float* hbuf = buf0;
    float* zbuf = buf1;
    const int n4    = (N_NODES * HDIM) / 4;
    const int cpBlk = (n4 + 255) / 256;
    const int agBlk = (N_EDGES + 7) / 8;                 // 8 wave32 (edges) per block
    for (int l = 0; l < 3; ++l) {
        void* const* cv = d_in + (2 + l * 10);
        const float* a1  = (const float*)cv[0];
        const float* a2  = (const float*)cv[1];
        const float* b1  = (const float*)cv[2];
        const float* b2  = (const float*)cv[3];
        const float* be1 = (const float*)cv[4];
        const float* bnb = (const float*)cv[5];
        const float* bng = (const float*)cv[6];
        const float* g1  = (const float*)cv[7];
        const float* w1  = (const float*)cv[8];
        const float* w2  = (const float*)cv[9];

        // z = h; z[dst] += h[src]  (L2-resident gather / scatter-add)
        k_copy_f4<<<cpBlk, 256, 0, stream>>>((float4*)zbuf, (const float4*)hbuf, n4);
        k_edge_agg<<<agBlk, 256, 0, stream>>>(hbuf, edges, zbuf, N_EDGES);
        // t = bn(prelu(z @ w1 + b1, a1), g1, be1)
        launch_gemm<HDIM, true, true>(zbuf, w1, b1, hbuf, N_NODES, HDIM,
                                      a1, g1, be1, stream);
        // h = bn(prelu(t @ w2 + b2, a2), bng, bnb)
        launch_gemm<HDIM, true, true>(hbuf, w2, b2, zbuf, N_NODES, HDIM,
                                      a2, bng, bnb, stream);
        float* t = hbuf; hbuf = zbuf; zbuf = t;          // new h in (old) zbuf
    }

    // 3) pooled graph embedding + heads
    k_zero<<<(NBATCH * HDIM + 255) / 256, 256, 0, stream>>>(gbuf, NBATCH * HDIM);
    k_pool<<<(N_NODES + 7) / 8, 256, 0, stream>>>(hbuf, batch, gbuf, N_NODES);
    k_bn_rows<<<(NBATCH * HDIM + 255) / 256, 256, 0, stream>>>(gbuf, bn_g, bn_b,
                                                               NBATCH * HDIM, HDIM);
    launch_gemm<HDIM, false, false>(gbuf, fc_w, fc_b, g2, NBATCH, HDIM,
                                    nullptr, nullptr, nullptr, stream);
    launch_gemm<HDIM, false, false>(g2, mu_w, mu_b, out_mu, NBATCH, LATD,
                                    nullptr, nullptr, nullptr, stream);
    launch_gemm<HDIM, false, false>(g2, lv_w, lv_b, out_lv, NBATCH, LATD,
                                    nullptr, nullptr, nullptr, stream);
    k_reparam<<<(NBATCH * LATD + 255) / 256, 256, 0, stream>>>(out_mu, out_lv, eps, out_xg,
                                                               NBATCH * LATD);

    // 4) decoder MLP
    launch_gemm<LATD, true, false>(out_xg, dec0_w, dec0_b, dbuf1, NBATCH, HDEC,
                                   dec_a, nullptr, nullptr, stream);
    launch_gemm<HDEC, true, false>(dbuf1, dec1_w, dec1_b, dbuf2, NBATCH, HDEC,
                                   dec_a, nullptr, nullptr, stream);
    launch_gemm<HDEC, false, false>(dbuf2, dec2_w, dec2_b, logit, NBATCH, NOUTV,
                                    nullptr, nullptr, nullptr, stream);

    // 5) Gumbel argmax -> symmetric adjacency
    k_adj<<<(NBATCH * NMAXV * NMAXV + 255) / 256, 256, 0, stream>>>(logit, gumbel, out_adj);
}
